// RegionTransformer_83107617177694
// MI455X (gfx1250) — matrix operand order repaired and checked by reading, not compiled, not run on hardware
//
#include <hip/hip_runtime.h>

typedef __attribute__((ext_vector_type(16))) _Float16 v16h;
typedef __attribute__((ext_vector_type(8)))  float    v8f;

#define WMMA_F16(A,B,C) __builtin_amdgcn_wmma_f32_16x16x32_f16(false,(A),false,(B),(short)0,(C),false,false)

// ---------------------------------------------------------------------------
// WMMA fragment loaders (contiguous k order used by this file, not AMD's table order; wave32).
// A 16x32 f16: lanes 0-15 row M=lane, elems 0-15 = K0-15;
//              lanes 16-31 same rows, elems 0-15 = K16-31.
// B 32x16 f16: col = lane&15; lanes 0-15 hold K0-15, lanes 16-31 hold K16-31.
//              All B sources are stored K-contiguous per column (transposed),
//              so each lane reads 32 contiguous bytes -> b128 loads.
// C/D f32:     VGPR r -> M = r (+8 for lanes 16-31), N = lane&15.
// ---------------------------------------------------------------------------
__device__ __forceinline__ v16h ldsA_frag(const _Float16* As, int ld, int mBase, int kBase, int lane) {
  v16h a;
  const _Float16* p = As + (mBase + (lane & 15)) * ld + kBase + ((lane >= 16) ? 16 : 0);
#pragma unroll
  for (int e = 0; e < 16; ++e) a[e] = p[e];
  return a;
}

// Bt layout: Bt[n * ldk + k]  (K contiguous per output column)
__device__ __forceinline__ v16h ldsBt_frag(const _Float16* Bt, int ldk, int kBase, int nBase, int lane) {
  v16h b;
  const _Float16* p = Bt + (nBase + (lane & 15)) * ldk + kBase + ((lane >= 16) ? 16 : 0);
#pragma unroll
  for (int e = 0; e < 16; ++e) b[e] = p[e];
  return b;
}

__device__ __forceinline__ v16h gBt_frag(const _Float16* __restrict__ Wt, int ldk, int kBase, int nBase, int lane) {
  v16h b;
  const _Float16* p = Wt + (size_t)(nBase + (lane & 15)) * ldk + kBase + ((lane >= 16) ? 16 : 0);
#pragma unroll
  for (int e = 0; e < 16; ++e) b[e] = p[e];
  return b;
}

// ---------------------------------------------------------------------------
// Weight transpose + f16 convert:  Wt[n*K + k] = (f16) W[k*N + n]
// ---------------------------------------------------------------------------
__global__ void transpose_f16(const float* __restrict__ W, _Float16* __restrict__ Wt,
                              int K, int N) {
  int i = blockIdx.x * blockDim.x + threadIdx.x;
  if (i >= K * N) return;
  int k = i / N, n = i - k * N;
  Wt[(size_t)n * K + k] = (_Float16)W[i];
}

// ---------------------------------------------------------------------------
// Generic GEMM: C[M,N] = A[M,K] @ W[K,N] (+bias, optional relu), f16 WMMA.
// Requires M%128==0, N%64==0, K%32==0 (true for all call sites).
// ---------------------------------------------------------------------------
__global__ __launch_bounds__(256) void gemm_f16wmma(
    const float* __restrict__ A, const float* __restrict__ W,
    const float* __restrict__ bias, float* __restrict__ C,
    int M, int K, int N, int relu)
{
  __shared__ _Float16 As[128 * 32];
  __shared__ _Float16 Bs[64 * 32];   // transposed: Bs[n*32 + k]
  int m0 = blockIdx.x * 128, n0 = blockIdx.y * 64;
  int tid = threadIdx.x, wid = tid >> 5, lane = tid & 31;
  int wm = (wid >> 1) * 32, wn = (wid & 1) * 32;
  v8f acc00 = {}, acc01 = {}, acc10 = {}, acc11 = {};
  for (int k0 = 0; k0 < K; k0 += 32) {
    for (int u = tid; u < 128 * 32; u += 256) {
      int r = u >> 5, kk = u & 31;
      As[u] = (_Float16)A[(size_t)(m0 + r) * K + k0 + kk];
    }
    if (k0 + 32 < K) __builtin_prefetch(&A[(size_t)(m0 + (tid >> 1)) * K + k0 + 32], 0, 1);
    for (int u = tid; u < 32 * 64; u += 256) {
      int kk = u >> 6, n = u & 63;                       // coalesced global read
      Bs[n * 32 + kk] = (_Float16)W[(size_t)(k0 + kk) * N + n0 + n];
    }
    __syncthreads();
    v16h a0 = ldsA_frag(As, 32, wm, 0, lane);
    v16h a1 = ldsA_frag(As, 32, wm + 16, 0, lane);
    v16h b0 = ldsBt_frag(Bs, 32, 0, wn, lane);
    v16h b1 = ldsBt_frag(Bs, 32, 0, wn + 16, lane);
    acc00 = WMMA_F16(a0, b0, acc00);
    acc01 = WMMA_F16(a0, b1, acc01);
    acc10 = WMMA_F16(a1, b0, acc10);
    acc11 = WMMA_F16(a1, b1, acc11);
    __syncthreads();
  }
  int mo = (lane >= 16) ? 8 : 0, lc = lane & 15;
  v8f accs[2][2] = {{acc00, acc01}, {acc10, acc11}};
#pragma unroll
  for (int i = 0; i < 2; ++i)
#pragma unroll
    for (int j = 0; j < 2; ++j) {
      int col = n0 + wn + j * 16 + lc;
      float bv = bias ? bias[col] : 0.f;
#pragma unroll
      for (int rr = 0; rr < 8; ++rr) {
        int row = m0 + wm + i * 16 + mo + rr;
        float v = accs[i][j][rr] + bv;
        if (relu) v = fmaxf(v, 0.f);
        C[(size_t)row * N + col] = v;
      }
    }
}

// ---------------------------------------------------------------------------
// Fused attention MLP: rows r in [0, 131072) over (bi, kk).
// A row = q[bi] - k[gather(idx)] + pe[r]  (gathered on the fly into LDS f16).
// SIM = relu(A @ W1 + b1) @ W2 + b2, hidden (64 x 4c) kept f16 in LDS.
// W1t layout: [4c][c] f16; W2t layout: [c][4c] f16 (K contiguous).
// ---------------------------------------------------------------------------
__global__ __launch_bounds__(256) void attn_mlp_fused(
    const float* __restrict__ QKV, const float* __restrict__ PE,
    const int* __restrict__ idx,
    const _Float16* __restrict__ W1t, const float* __restrict__ b1,
    const _Float16* __restrict__ W2t, const float* __restrict__ b2,
    float* __restrict__ SIM, int c)
{
  extern __shared__ _Float16 smf[];
  _Float16* As = smf;             // 64 x 128 (stride 128)
  _Float16* Hs = smf + 64 * 128;  // 64 x hid (stride hid)
  int hid = 4 * c, c3 = 3 * c;
  int r0 = blockIdx.x * 64;
  int tid = threadIdx.x, wid = tid >> 5, lane = tid & 31;

  for (int u = tid; u < 64 * c; u += 256) {
    int rr = u / c, ch = u - rr * c;
    int r = r0 + rr, bi = r >> 4, kk = r & 15;
    int j = ((bi >> 10) << 10) + idx[bi * 16 + kk];
    float av = QKV[(size_t)bi * c3 + ch] - QKV[(size_t)j * c3 + c + ch]
             + PE[(size_t)r * c + ch];
    As[rr * 128 + ch] = (_Float16)av;
  }
  __syncthreads();

  // phase 1: H = relu(A @ W1 + b1)   (each wave: round-robin 16-wide strips)
  for (int nt = wid * 16; nt < hid; nt += 8 * 16) {
    v8f a0 = {}, a1 = {}, a2 = {}, a3 = {};
    for (int kt = 0; kt < c; kt += 32) {
      v16h bf = gBt_frag(W1t, c, kt, nt, lane);
      v16h f0 = ldsA_frag(As, 128, 0,  kt, lane);
      v16h f1 = ldsA_frag(As, 128, 16, kt, lane);
      v16h f2 = ldsA_frag(As, 128, 32, kt, lane);
      v16h f3 = ldsA_frag(As, 128, 48, kt, lane);
      a0 = WMMA_F16(f0, bf, a0);
      a1 = WMMA_F16(f1, bf, a1);
      a2 = WMMA_F16(f2, bf, a2);
      a3 = WMMA_F16(f3, bf, a3);
    }
    int mo = (lane >= 16) ? 8 : 0, col = nt + (lane & 15);
    float bv = b1[col];
    v8f av[4] = {a0, a1, a2, a3};
#pragma unroll
    for (int mt = 0; mt < 4; ++mt)
#pragma unroll
      for (int rr = 0; rr < 8; ++rr) {
        float v = fmaxf(av[mt][rr] + bv, 0.f);
        Hs[(mt * 16 + mo + rr) * hid + col] = (_Float16)v;
      }
  }
  __syncthreads();

  // phase 2: SIM = H @ W2 + b2
  int ntile = c >> 4;
  for (int t = wid; t < 4 * ntile; t += 8) {
    int mt = t / ntile, nt = (t - mt * ntile) << 4;
    v8f acc = {};
    for (int kt = 0; kt < hid; kt += 32) {
      v16h bf = gBt_frag(W2t, hid, kt, nt, lane);
      v16h af = ldsA_frag(Hs, hid, mt * 16, kt, lane);
      acc = WMMA_F16(af, bf, acc);
    }
    int mo = (lane >= 16) ? 8 : 0, col = nt + (lane & 15);
    float bv = b2[col];
#pragma unroll
    for (int rr = 0; rr < 8; ++rr)
      SIM[(size_t)(r0 + mt * 16 + mo + rr) * c + col] = acc[rr] + bv;
  }
}

// ---------------------------------------------------------------------------
// Glue kernels (bandwidth-trivial)
// ---------------------------------------------------------------------------
__global__ void embed7(const float* __restrict__ in, const float* __restrict__ W,
                       const float* __restrict__ b, float* __restrict__ X) {
  int i = blockIdx.x * blockDim.x + threadIdx.x;
  if (i >= 8192 * 64) return;
  int row = i >> 6, ch = i & 63;
  const float* p = in + (size_t)row * 10 + 3;
  float s = b[ch];
#pragma unroll
  for (int f = 0; f < 7; ++f) s += p[f] * W[f * 64 + ch];
  X[i] = s;
}

__global__ void topk16(const float* __restrict__ pts, int* __restrict__ IDX) {
  int bi = blockIdx.x * blockDim.x + threadIdx.x;
  if (bi >= 8192) return;
  int b0 = (bi >> 10) << 10;
  const float* pi = pts + (size_t)bi * 10;
  float px = pi[0], py = pi[1], pz = pi[2];
  float bd[16]; int bj[16];
#pragma unroll
  for (int k = 0; k < 16; ++k) { bd[k] = 3.0e38f; bj[k] = 0; }
  for (int j = 0; j < 1024; ++j) {
    const float* pj = pts + (size_t)(b0 + j) * 10;
    float dx = px - pj[0], dy = py - pj[1], dz = pz - pj[2];
    float d = dx * dx + dy * dy + dz * dz;
    if (d < bd[15]) {
      int p = 15;
      while (p > 0 && bd[p - 1] > d) { bd[p] = bd[p - 1]; bj[p] = bj[p - 1]; --p; }
      bd[p] = d; bj[p] = j;
    }
  }
#pragma unroll
  for (int k = 0; k < 16; ++k) IDX[bi * 16 + k] = bj[k];
}

__global__ __launch_bounds__(256) void pos_mlp(
    const float* __restrict__ pts, const int* __restrict__ idx,
    const float* __restrict__ w1, const float* __restrict__ b1,
    const float* __restrict__ w2, const float* __restrict__ b2,
    float* __restrict__ PE, int c)
{
  __shared__ float hs[32 * 32];
  int tid = threadIdx.x;
  int r0 = blockIdx.x * 32;
  for (int u = tid; u < 32 * 32; u += 256) {
    int rr = u >> 5, h = u & 31;
    int r = r0 + rr, bi = r >> 4, kk = r & 15;
    int j = ((bi >> 10) << 10) + idx[bi * 16 + kk];
    const float* pi = pts + (size_t)bi * 10;
    const float* pj = pts + (size_t)j * 10;
    float r0v = pi[0] - pj[0], r1v = pi[1] - pj[1], r2v = pi[2] - pj[2];
    float v = r0v * w1[0 * 32 + h] + r1v * w1[1 * 32 + h] + r2v * w1[2 * 32 + h] + b1[h];
    hs[u] = fmaxf(v, 0.f);
  }
  __syncthreads();
  for (int u = tid; u < 32 * c; u += 256) {
    int rr = u / c, ch = u - rr * c;
    float s = b2[ch];
#pragma unroll
    for (int h = 0; h < 32; ++h) s += hs[rr * 32 + h] * w2[h * c + ch];
    PE[(size_t)(r0 + rr) * c + ch] = s;
  }
}

__global__ void attn_softmax(const float* __restrict__ SIM, const float* __restrict__ QKV,
                             const float* __restrict__ PE, const int* __restrict__ idx,
                             float* __restrict__ T, int c)
{
  int i = blockIdx.x * blockDim.x + threadIdx.x;
  if (i >= 8192 * c) return;
  int bi = i / c, ch = i - bi * c;
  int c3 = 3 * c, b0 = (bi >> 10) << 10;
  float mx = -3.0e38f; float sv[16];
#pragma unroll
  for (int kk = 0; kk < 16; ++kk) {
    float s = SIM[(size_t)(bi * 16 + kk) * c + ch];
    sv[kk] = s; mx = fmaxf(mx, s);
  }
  float se = 0.f, acc = 0.f;
#pragma unroll
  for (int kk = 0; kk < 16; ++kk) {
    float e = __expf(sv[kk] - mx); se += e;
    int j = b0 + idx[bi * 16 + kk];
    float vj = QKV[(size_t)j * c3 + 2 * c + ch] + PE[(size_t)(bi * 16 + kk) * c + ch];
    acc += e * vj;
  }
  T[i] = acc / se;
}

__global__ __launch_bounds__(256) void bn_stats(const float* __restrict__ X, int C, int rows,
                                                float* __restrict__ stats) {
  int ch = blockIdx.x, tid = threadIdx.x;
  float s = 0.f, q = 0.f;
  for (int r = tid; r < rows; r += 256) {
    float v = X[(size_t)r * C + ch]; s += v; q += v * v;
  }
  __shared__ float sh[256], sh2[256];
  sh[tid] = s; sh2[tid] = q; __syncthreads();
  for (int o = 128; o > 0; o >>= 1) {
    if (tid < o) { sh[tid] += sh[tid + o]; sh2[tid] += sh2[tid + o]; }
    __syncthreads();
  }
  if (tid == 0) {
    float m = sh[0] / rows, var = sh2[0] / rows - m * m;
    stats[ch] = m; stats[C + ch] = rsqrtf(var + 1e-5f);
  }
}

__global__ void bn_apply(const float* __restrict__ X, const float* __restrict__ stats,
                         const float* __restrict__ g, const float* __restrict__ b,
                         float* __restrict__ Y, int C, int n, int relu) {
  int i = blockIdx.x * blockDim.x + threadIdx.x;
  if (i >= n) return;
  int ch = i % C;
  float v = (X[i] - stats[ch]) * stats[C + ch] * g[ch] + b[ch];
  if (relu) v = fmaxf(v, 0.f);
  Y[i] = v;
}

__global__ void add_vec(float* __restrict__ Y, const float* __restrict__ X, int n) {
  int i = blockIdx.x * blockDim.x + threadIdx.x;
  if (i < n) Y[i] += X[i];
}

__global__ void maxpool2(const float* __restrict__ XI, const float* __restrict__ XN,
                         float* __restrict__ G) {
  int i = blockIdx.x * blockDim.x + threadIdx.x;
  if (i >= 2048) return;
  int b = i >> 8, gc = i & 255;
  const float* src = (gc < 128) ? XI : XN;
  int ch = gc & 127;
  float m = -3.0e38f;
  for (int r = 0; r < 1024; ++r) m = fmaxf(m, src[(size_t)((b << 10) + r) * 128 + ch]);
  G[i] = m;
}

__global__ void concat_g_res(const float* __restrict__ G, const float* __restrict__ RES,
                             float* __restrict__ CAT) {
  int i = blockIdx.x * blockDim.x + threadIdx.x;
  if (i >= 8192 * 320) return;
  int row = i / 320, col = i - row * 320;
  int b = row >> 10;
  CAT[i] = (col < 256) ? G[b * 256 + col] : RES[(size_t)row * 64 + (col - 256)];
}

__global__ void head1(const float* __restrict__ X, const float* __restrict__ w,
                      const float* __restrict__ b, float* __restrict__ out) {
  int row = blockIdx.x * blockDim.x + threadIdx.x;
  if (row >= 8192) return;
  const float* p = X + (size_t)row * 128;
  float s = b[0];
#pragma unroll
  for (int ch = 0; ch < 128; ++ch) s += p[ch] * w[ch];
  out[row] = s;
}

// ---------------------------------------------------------------------------
// Host orchestration
// ---------------------------------------------------------------------------
struct BP {
  const float *attn_b1, *attn_b2, *attn_w1, *attn_w2;
  const float *bn1_b, *bn1_g, *bn2_b, *bn2_g, *bn3_b, *bn3_g;
  const float *fc1_b, *fc1_w, *fc2_b, *fc2_w;
  const float *pos_b1, *pos_b2, *pos_w1, *pos_w2, *qkv_w;
};

static BP mk_bp(void* const* d_in, int b) {
  BP p;
  p.attn_b1 = (const float*)d_in[b + 0];  p.attn_b2 = (const float*)d_in[b + 1];
  p.attn_w1 = (const float*)d_in[b + 2];  p.attn_w2 = (const float*)d_in[b + 3];
  p.bn1_b   = (const float*)d_in[b + 4];  p.bn1_g   = (const float*)d_in[b + 5];
  p.bn2_b   = (const float*)d_in[b + 6];  p.bn2_g   = (const float*)d_in[b + 7];
  p.bn3_b   = (const float*)d_in[b + 8];  p.bn3_g   = (const float*)d_in[b + 9];
  p.fc1_b   = (const float*)d_in[b + 10]; p.fc1_w   = (const float*)d_in[b + 11];
  p.fc2_b   = (const float*)d_in[b + 12]; p.fc2_w   = (const float*)d_in[b + 13];
  p.pos_b1  = (const float*)d_in[b + 14]; p.pos_b2  = (const float*)d_in[b + 15];
  p.pos_w1  = (const float*)d_in[b + 16]; p.pos_w2  = (const float*)d_in[b + 17];
  p.qkv_w   = (const float*)d_in[b + 18];
  return p;
}

static void run_block(const float* x_in, int cin, int cout, float* x_out,
                      const float* pts, const int* idx, const BP& p,
                      float* PRE, float* H, float* T, float* QKV,
                      float* PE, float* SIM, float* STATS,
                      _Float16* W1T, _Float16* W2T, hipStream_t s)
{
  const int M = 8192;
  const int n1 = M * cout;
  const int hid = 4 * cout;
  dim3 blk(256);
  gemm_f16wmma<<<dim3(M / 128, cout / 64), blk, 0, s>>>(x_in, p.fc1_w, p.fc1_b, PRE, M, cin, cout, 0);
  bn_stats<<<cout, blk, 0, s>>>(PRE, cout, M, STATS);
  bn_apply<<<(n1 + 255) / 256, blk, 0, s>>>(PRE, STATS, p.bn1_g, p.bn1_b, H, cout, n1, 1);
  gemm_f16wmma<<<dim3(M / 128, (3 * cout) / 64), blk, 0, s>>>(H, p.qkv_w, nullptr, QKV, M, cout, 3 * cout, 0);
  pos_mlp<<<131072 / 32, blk, 0, s>>>(pts, idx, p.pos_w1, p.pos_b1, p.pos_w2, p.pos_b2, PE, cout);
  transpose_f16<<<(cout * hid + 255) / 256, blk, 0, s>>>(p.attn_w1, W1T, cout, hid);
  transpose_f16<<<(hid * cout + 255) / 256, blk, 0, s>>>(p.attn_w2, W2T, hid, cout);
  size_t smb = ((size_t)64 * 128 + (size_t)64 * hid) * sizeof(_Float16);
  attn_mlp_fused<<<131072 / 64, blk, smb, s>>>(QKV, PE, idx, W1T, p.attn_b1, W2T, p.attn_b2, SIM, cout);
  attn_softmax<<<(n1 + 255) / 256, blk, 0, s>>>(SIM, QKV, PE, idx, T, cout);
  bn_stats<<<cout, blk, 0, s>>>(T, cout, M, STATS);
  bn_apply<<<(n1 + 255) / 256, blk, 0, s>>>(T, STATS, p.bn2_g, p.bn2_b, PRE, cout, n1, 1);
  add_vec<<<(n1 + 255) / 256, blk, 0, s>>>(PRE, H, n1);
  gemm_f16wmma<<<dim3(M / 128, cout / 64), blk, 0, s>>>(PRE, p.fc2_w, p.fc2_b, T, M, cout, cout, 0);
  bn_stats<<<cout, blk, 0, s>>>(T, cout, M, STATS);
  bn_apply<<<(n1 + 255) / 256, blk, 0, s>>>(T, STATS, p.bn3_g, p.bn3_b, x_out, cout, n1, 1);
}

extern "C" void kernel_launch(void* const* d_in, const int* in_sizes, int n_in,
                              void* d_out, int out_size, void* d_ws, size_t ws_size,
                              hipStream_t stream) {
  (void)in_sizes; (void)out_size; (void)ws_size;
  if (n_in < 162) return;
  const float* inl = (const float*)d_in[0];
  const float* nbr = (const float*)d_in[1];
  const float* am_b = (const float*)d_in[2];
  const float* am_w = (const float*)d_in[3];
  BP b1i = mk_bp(d_in, 4),   b1n = mk_bp(d_in, 23);
  BP b2i0 = mk_bp(d_in, 42), b2i1 = mk_bp(d_in, 61);
  BP b2n0 = mk_bp(d_in, 80), b2n1 = mk_bp(d_in, 99);
  BP b3i = mk_bp(d_in, 118), b3n = mk_bp(d_in, 137);
  const float* fc1_b  = (const float*)d_in[156];
  const float* fc1_w  = (const float*)d_in[157];
  const float* nfc1_b = (const float*)d_in[158];
  const float* nfc1_w = (const float*)d_in[159];
  const float* rm_b   = (const float*)d_in[160];
  const float* rm_w   = (const float*)d_in[161];

  char* cur = (char*)d_ws;
  auto carve = [&](size_t bytes) -> char* {
    char* p = cur; cur += (bytes + 255) & ~(size_t)255; return p;
  };
  int*      IDX_I = (int*)carve(8192 * 16 * sizeof(int));
  int*      IDX_N = (int*)carve(8192 * 16 * sizeof(int));
  float*    XI    = (float*)carve((size_t)8192 * 128 * 4);
  float*    XN    = (float*)carve((size_t)8192 * 128 * 4);
  float*    RES   = (float*)carve((size_t)8192 * 64 * 4);
  float*    RESN  = (float*)carve((size_t)8192 * 64 * 4);
  float*    CAT   = (float*)carve((size_t)8192 * 320 * 4);
  float*    PRE   = (float*)carve((size_t)8192 * 128 * 4);
  float*    H     = (float*)carve((size_t)8192 * 128 * 4);
  float*    T     = (float*)carve((size_t)8192 * 128 * 4);
  float*    QKV   = (float*)carve((size_t)8192 * 384 * 4);
  float*    G     = (float*)carve((size_t)8 * 256 * 4);
  float*    STATS = (float*)carve((size_t)512 * 4);
  _Float16* W1T   = (_Float16*)carve((size_t)128 * 512 * 2);
  _Float16* W2T   = (_Float16*)carve((size_t)512 * 128 * 2);
  float*    PE    = (float*)carve((size_t)131072 * 128 * 4);
  float*    SIM   = (float*)carve((size_t)131072 * 128 * 4);

  dim3 blk(256);
  embed7<<<(8192 * 64 + 255) / 256, blk, 0, stream>>>(inl, fc1_w, fc1_b, XI);
  embed7<<<(8192 * 64 + 255) / 256, blk, 0, stream>>>(nbr, nfc1_w, nfc1_b, XN);
  topk16<<<8192 / 256, blk, 0, stream>>>(inl, IDX_I);
  topk16<<<8192 / 256, blk, 0, stream>>>(nbr, IDX_N);

  run_block(XI, 64, 64, RES,  inl, IDX_I, b1i,  PRE, H, T, QKV, PE, SIM, STATS, W1T, W2T, stream);
  run_block(XN, 64, 64, RESN, nbr, IDX_N, b1n,  PRE, H, T, QKV, PE, SIM, STATS, W1T, W2T, stream);
  run_block(RES,  64, 128, XI, inl, IDX_I, b2i0, PRE, H, T, QKV, PE, SIM, STATS, W1T, W2T, stream);
  run_block(RESN, 64, 128, XN, nbr, IDX_N, b2n0, PRE, H, T, QKV, PE, SIM, STATS, W1T, W2T, stream);
  run_block(XI, 128, 128, XI, inl, IDX_I, b2i1, PRE, H, T, QKV, PE, SIM, STATS, W1T, W2T, stream);
  run_block(XN, 128, 128, XN, nbr, IDX_N, b2n1, PRE, H, T, QKV, PE, SIM, STATS, W1T, W2T, stream);

  maxpool2<<<(2048 + 255) / 256, blk, 0, stream>>>(XI, XN, G);
  concat_g_res<<<(8192 * 320 + 255) / 256, blk, 0, stream>>>(G, RES, CAT);
  run_block(CAT, 320, 128, XI, inl, IDX_I, b3i, PRE, H, T, QKV, PE, SIM, STATS, W1T, W2T, stream);
  concat_g_res<<<(8192 * 320 + 255) / 256, blk, 0, stream>>>(G, RESN, CAT);
  run_block(CAT, 320, 128, XN, nbr, IDX_N, b3n, PRE, H, T, QKV, PE, SIM, STATS, W1T, W2T, stream);

  float* out = (float*)d_out;
  head1<<<8192 / 256, blk, 0, stream>>>(XI, rm_w, rm_b, out);
  head1<<<8192 / 256, blk, 0, stream>>>(XN, am_w, am_b, out + 8192);
}
